// STAR_76209899700547
// MI455X (gfx1250) — matrix operand order repaired and checked by reading, not compiled, not run on hardware
//
#include <hip/hip_runtime.h>
#include <hip/hip_bf16.h>

// MI455X / gfx1250: wave32, v_wmma_f32_16x16x32_f16.
// Routed 2-branch MLP: out = route(relu(x@W1f_b + b1f_b)@W2f_b + b2f_b,
//                                  relu(x@W1f_c + b1f_c)@W2f_c + b2f_c)
// Roofline: 512 MB obs+out @ 23.3 TB/s ~= 22 us; f16 WMMA for both branches
// (69 GFLOP) lands at the same point -> spill-free WMMA, packed-f16 epilogue,
// streaming b128 loads/stores, minimal LDS traffic.

typedef __attribute__((ext_vector_type(16))) _Float16 v16h;
typedef __attribute__((ext_vector_type(8)))  _Float16 v8h;
typedef __attribute__((ext_vector_type(8)))  float    v8f;
typedef __attribute__((ext_vector_type(4)))  float    v4f;

#define NROWS   (1 << 20)
#define IN_DIM  64
#define HID     128
#define FEAT    64
#define MT      2                         // row-tiles (of 16) per wave iteration
#define NPAIRS  (NROWS / (16 * MT))       // 32768
#define WAVES_PER_WG 8
#define NWG     512

__global__ __launch_bounds__(256)
void STAR_moe_mlp_wmma4(const float* __restrict__ obs,
                        const float* __restrict__ W1b, const float* __restrict__ b1b,
                        const float* __restrict__ W2b, const float* __restrict__ b2b,
                        const float* __restrict__ W1c, const float* __restrict__ b1c,
                        const float* __restrict__ W2c, const float* __restrict__ b2c,
                        const float* __restrict__ W1p, const float* __restrict__ b1p,
                        const float* __restrict__ W2p, const float* __restrict__ b2p,
                        float* __restrict__ out) {
  alignas(32) __shared__ _Float16 sW1[2][IN_DIM * HID];          // 32 KB  [K][N]
  alignas(32) __shared__ _Float16 sW2T[2][FEAT * HID];           // 32 KB  [FEAT][HID]
  alignas(16) __shared__ float    sB1[2][HID];                   // 1 KB
  alignas(16) __shared__ float    sB2[2][FEAT];                  // 0.5 KB
  alignas(32) __shared__ _Float16 sHT[WAVES_PER_WG][MT * HID * 16]; // h^T, 64 KB

  const int tid = threadIdx.x;

  // ---- fuse weights into LDS (f16) once per WG; weights are L2-resident ----
  for (int i = tid; i < IN_DIM * HID; i += 256) {
    float p = W1p[i];
    sW1[0][i] = (_Float16)(W1b[i] * p);
    sW1[1][i] = (_Float16)(W1c[i] * p);
  }
  for (int i = tid; i < HID * FEAT; i += 256) {
    int h = i >> 6, f = i & 63;           // W2 row-major [HID][FEAT]
    float p = W2p[i];
    sW2T[0][f * HID + h] = (_Float16)(W2b[i] * p);
    sW2T[1][f * HID + h] = (_Float16)(W2c[i] * p);
  }
  for (int i = tid; i < HID; i += 256) {
    sB1[0][i] = b1b[i] + b1p[i];
    sB1[1][i] = b1c[i] + b1p[i];
  }
  if (tid < FEAT) {
    sB2[0][tid] = b2b[tid] + b2p[tid];
    sB2[1][tid] = b2c[tid] + b2p[tid];
  }
  __syncthreads();

  const int lane  = tid & 31;
  const int wave  = tid >> 5;
  const int lrow  = lane & 15;
  const int lhalf = (lane >> 4) & 1;
  _Float16* hT0 = sHT[wave];

  const int gwave  = blockIdx.x * WAVES_PER_WG + wave;
  const int nwaves = gridDim.x * WAVES_PER_WG;

  const v8h zero8h = {};

  // Wave-uniform tile loop => EXEC all-ones around every WMMA.
  for (int t = gwave; t < NPAIRS; t += nwaves) {
    const int row0 = t * (16 * MT);

    // ---- x A-fragments (16x32 f16 layout, ISA 7.12.2) for MT row-tiles ----
    v16h xa[MT][2];
    bool isb[MT];
#pragma unroll
    for (int u = 0; u < MT; ++u) {
      const float* xrow = obs + (size_t)(row0 + u * 16 + lrow) * IN_DIM;
#pragma unroll
      for (int kc = 0; kc < 2; ++kc) {
        const v4f* p4 = (const v4f*)(xrow + kc * 32 + lhalf * 8);
        v4f f0 = p4[0], f1 = p4[1], f2 = p4[4], f3 = p4[5];
#pragma unroll
        for (int j = 0; j < 4; ++j) {
          xa[u][kc][j]      = (_Float16)f0[j];   // K = kc*32 + lhalf*8 + j
          xa[u][kc][4 + j]  = (_Float16)f1[j];
          xa[u][kc][8 + j]  = (_Float16)f2[j];   // K = ... + 16 + j
          xa[u][kc][12 + j] = (_Float16)f3[j];
        }
      }
      isb[u] = (xrow[2] == 1.0f) && (xrow[3] == 0.0f);
    }

#pragma unroll
    for (int net = 0; net < 2; ++net) {
      const _Float16* w1  = sW1[net];
      const _Float16* w2t = sW2T[net];

      // ---- layer 1: h = x @ W1f + b1f; epilogue: cvt-pk then packed-f16 ReLU ----
#pragma unroll
      for (int n = 0; n < 8; ++n) {
        const float bias = sB1[net][n * 16 + lrow];   // C-frag lane -> col
        v8f acc[MT];
#pragma unroll
        for (int u = 0; u < MT; ++u)
#pragma unroll
          for (int r = 0; r < 8; ++r) acc[u][r] = bias;
#pragma unroll
        for (int kc = 0; kc < 2; ++kc) {
          const _Float16* bp = w1 + (kc * 32 + lhalf * 8) * HID + n * 16 + lrow;  // B-frag lane -> N
          v16h bf;
#pragma unroll
          for (int j = 0; j < 8; ++j) {
            bf[j]     = bp[j * HID];
            bf[8 + j] = bp[(16 + j) * HID];
          }
#pragma unroll
          for (int u = 0; u < MT; ++u)                // reuse bf for both tiles
            acc[u] = __builtin_amdgcn_wmma_f32_16x16x32_f16(
                false, xa[u][kc], false, bf, (short)0, acc[u], false, false);
        }
#pragma unroll
        for (int u = 0; u < MT; ++u) {
          // relu(cvt(h)) == cvt(relu(h)): 4x v_cvt_pk_f16_f32 + 4x v_pk_max_num_f16
          v8h hh;
#pragma unroll
          for (int r = 0; r < 8; ++r) hh[r] = (_Float16)acc[u][r];
          hh = __builtin_elementwise_max(hh, zero8h);
          // C-frag: lane -> col n*16+lrow, vgpr r -> row lhalf*8+r
          *(v8h*)(hT0 + u * (HID * 16) + (n * 16 + lrow) * 16 + lhalf * 8) = hh;
        }
      }
      // same-wave LDS RAW: DS is in-order per wave, no barrier needed

      // ---- layer 2 transposed: out^T = W2T(64x128) @ h^T(128x16) ----
      // kc-outer / mt-inner: each h^T B-frag loaded once per kc (4x fewer
      // B-frag DS loads); all four mt accumulators stay live (fits <256 VGPRs,
      // occupancy is LDS-limited anyway).
      v8f acc[4][MT];
#pragma unroll
      for (int mt = 0; mt < 4; ++mt)
#pragma unroll
        for (int r = 0; r < 8; ++r) {
          const float b2v = sB2[net][mt * 16 + lhalf * 8 + r];
#pragma unroll
          for (int u = 0; u < MT; ++u) acc[mt][u][r] = b2v;
        }
#pragma unroll
      for (int kc = 0; kc < 4; ++kc) {
        v16h bf[MT];
#pragma unroll
        for (int u = 0; u < MT; ++u)
#pragma unroll
          for (int j = 0; j < 8; ++j) {
            bf[u][j]     = hT0[u * (HID * 16) + (kc * 32 + lhalf * 8 + j) * 16 + lrow];
            bf[u][8 + j] = hT0[u * (HID * 16) + (kc * 32 + 16 + lhalf * 8 + j) * 16 + lrow];
          }
#pragma unroll
        for (int mt = 0; mt < 4; ++mt) {
          const _Float16* ap = w2t + (mt * 16 + lrow) * HID + kc * 32 + lhalf * 8;
          v16h af;
#pragma unroll
          for (int j = 0; j < 8; ++j) {
            af[j]     = ap[j];
            af[8 + j] = ap[16 + j];
          }
#pragma unroll
          for (int u = 0; u < MT; ++u)
            acc[mt][u] = __builtin_amdgcn_wmma_f32_16x16x32_f16(
                false, af, false, bf[u], (short)0, acc[mt][u], false, false);
        }
      }
      // ---- routed, EXEC-masked stores: each row written exactly once ----
#pragma unroll
      for (int u = 0; u < MT; ++u) {
        const bool pred = (net == 0) ? isb[u] : !isb[u];
        if (pred) {
          float* orow = out + (size_t)(row0 + u * 16 + lrow) * FEAT + lhalf * 8;
#pragma unroll
          for (int mt = 0; mt < 4; ++mt) {
            v4f lo, hi;
#pragma unroll
            for (int j = 0; j < 4; ++j) {
              lo[j] = acc[mt][u][j];
              hi[j] = acc[mt][u][4 + j];
            }
            *(v4f*)(orow + mt * 16)     = lo;
            *(v4f*)(orow + mt * 16 + 4) = hi;
          }
        }
      }
    }
  }
}

extern "C" void kernel_launch(void* const* d_in, const int* in_sizes, int n_in,
                              void* d_out, int out_size, void* d_ws, size_t ws_size,
                              hipStream_t stream) {
  (void)in_sizes; (void)n_in; (void)d_ws; (void)ws_size; (void)out_size;
  const float* obs = (const float*)d_in[0];
  const float* W1b = (const float*)d_in[1];
  const float* b1b = (const float*)d_in[2];
  const float* W2b = (const float*)d_in[3];
  const float* b2b = (const float*)d_in[4];
  const float* W1c = (const float*)d_in[5];
  const float* b1c = (const float*)d_in[6];
  const float* W2c = (const float*)d_in[7];
  const float* b2c = (const float*)d_in[8];
  const float* W1p = (const float*)d_in[9];
  const float* b1p = (const float*)d_in[10];
  const float* W2p = (const float*)d_in[11];
  const float* b2p = (const float*)d_in[12];
  float* out = (float*)d_out;

  STAR_moe_mlp_wmma4<<<NWG, 256, 0, stream>>>(
      obs, W1b, b1b, W2b, b2b, W1c, b1c, W2c, b2c, W1p, b1p, W2p, b2p, out);
}
